// G_Align_14628658610465
// MI455X (gfx1250) — matrix operand order repaired and checked by reading, not compiled, not run on hardware
//
#include <hip/hip_runtime.h>
#include <hip/hip_bf16.h>
#include <hip/hip_fp16.h>

#define NN 50000
#define EE 800000
#define FF 128
#define NLAYERS 2

typedef __attribute__((ext_vector_type(16))) _Float16 v16h;
typedef __attribute__((ext_vector_type(8)))  _Float16 v8h;
typedef __attribute__((ext_vector_type(8)))  float    v8f;

__device__ __forceinline__ v8f vzero8() {
  v8f z;
#pragma unroll
  for (int i = 0; i < 8; ++i) z[i] = 0.0f;
  return z;
}

__device__ __forceinline__ v8f wmma_f16(v16h a, v16h b, v8f c) {
  // D = A(16x32 f16) * B(32x16 f16) + C(16x16 f32)
  return __builtin_amdgcn_wmma_f32_16x16x32_f16(
      /*neg_a=*/false, a, /*neg_b=*/false, b,
      /*c_mod=*/(short)0, c, /*reuse_a=*/false, /*reuse_b=*/false);
}

// A fragment (16x32) from row-major f16 p[row*ld + k], cols kBase..kBase+31.
// k order used here (same as fragB_nm): lane half hh holds K = kBase+16hh..+15 contiguous
// -> two contiguous 8-half (16B) vector loads.
__device__ __forceinline__ v16h fragA_rm(const _Float16* p, int ld, int kBase) {
  int lane = threadIdx.x & 31;
  int m = lane & 15, hh = lane >> 4;
  const _Float16* b0 = p + m * ld + kBase + 16 * hh;
  v8h lo = *(const v8h*)b0;
  v8h hi = *(const v8h*)(b0 + 8);
  v16h a;
#pragma unroll
  for (int i = 0; i < 8; ++i) { a[i] = lo[i]; a[8 + i] = hi[i]; }
  return a;
}

// B fragment (32x16) from N-major (transposed) f16 pt[n*ld + k].
// k order used here (same as fragA_rm): lane (n = lane&15, hh = lane>>4) holds K = kBase+16hh..+15 contiguous
// -> a single 16-half (32B) vector load.
__device__ __forceinline__ v16h fragB_nm(const _Float16* pt, int ld, int kBase, int nBase) {
  int lane = threadIdx.x & 31;
  int n = lane & 15, hh = lane >> 4;
  return *(const v16h*)(pt + (size_t)(nBase + n) * ld + kBase + 16 * hh);
}

// ---------------- utility kernels ----------------
__global__ void k_zero(float* p, int n) {
  int i = blockIdx.x * blockDim.x + threadIdx.x;
  if (i < n) p[i] = 0.0f;
}

// transpose-convert a 128x128 fp32 matrix into N-major f16: dst[c*128 + r] = src[r*128 + c]
__global__ void k_cvtT_128(const float* __restrict__ src, _Float16* __restrict__ dst) {
  int i = blockIdx.x * blockDim.x + threadIdx.x;
  if (i < FF * FF) {
    int r = i >> 7, c = i & 127;
    dst[c * FF + r] = (_Float16)src[i];
  }
}

__global__ void k_deg(const int* col, float* deg, int e_count) {
  int e = blockIdx.x * blockDim.x + threadIdx.x;
  if (e < e_count) atomicAdd(&deg[col[e]], 1.0f);
}

// ---------------- input projection + LN + relu ----------------
// grid = N/16 blocks, 256 threads (8 waves); wave w computes output cols [16w,16w+16)
__global__ void k_inproj(const float* __restrict__ feats, const _Float16* __restrict__ WT16,
                         const float* __restrict__ bias, const float* __restrict__ g,
                         const float* __restrict__ beta,
                         float* __restrict__ xf32, _Float16* __restrict__ xf16) {
  __shared__ __align__(32) _Float16 sx[16 * FF];
  __shared__ float sy[16 * FF];
  int node0 = blockIdx.x * 16;
  for (int i = threadIdx.x; i < 16 * FF; i += 256)
    sx[i] = (_Float16)feats[node0 * FF + i];
  __syncthreads();

  int w = threadIdx.x >> 5, n0 = w * 16;
  int lane = threadIdx.x & 31, n = lane & 15, hh = lane >> 4;
  v8f acc = vzero8();
#pragma unroll
  for (int kb = 0; kb < FF; kb += 32)
    acc = wmma_f16(fragA_rm(sx, FF, kb), fragB_nm(WT16, FF, kb, n0), acc);

  float bv = bias[n0 + n];
#pragma unroll
  for (int r = 0; r < 8; ++r)
    sy[(r + 8 * hh) * FF + n0 + n] = acc[r] + bv;
  __syncthreads();

  if (threadIdx.x < 16) {
    float* rowp = &sy[threadIdx.x * FF];
    float mu = 0.f;
    for (int i = 0; i < FF; ++i) mu += rowp[i];
    mu *= (1.0f / FF);
    float var = 0.f;
    for (int i = 0; i < FF; ++i) { float d = rowp[i] - mu; var += d * d; }
    var *= (1.0f / FF);
    float inv = rsqrtf(var + 1e-5f);
    int gn = node0 + threadIdx.x;
    for (int i = 0; i < FF; ++i) {
      float y = (rowp[i] - mu) * inv * g[i] + beta[i];
      y = y > 0.f ? y : 0.f;
      xf32[gn * FF + i] = y;
      xf16[gn * FF + i] = (_Float16)y;
    }
  }
}

// ---------------- q/k/v projection + reductions ----------------
__global__ void k_qkv(const _Float16* __restrict__ xf16,
                      const _Float16* __restrict__ WqT, const _Float16* __restrict__ WkT,
                      const _Float16* __restrict__ WvT,
                      const float* __restrict__ bq, const float* __restrict__ bk,
                      const float* __restrict__ bv,
                      _Float16* __restrict__ q16, _Float16* __restrict__ k16,
                      float* __restrict__ v32,
                      float* __restrict__ ksum, float* __restrict__ vsum,
                      float* __restrict__ qn2, float* __restrict__ kn2) {
  __shared__ __align__(32) _Float16 sx[16 * FF];
  __shared__ float sred[2];
  int node0 = blockIdx.x * 16;
  for (int i = threadIdx.x; i < 16 * FF; i += 256)
    sx[i] = xf16[node0 * FF + i];
  if (threadIdx.x < 2) sred[threadIdx.x] = 0.f;
  __syncthreads();

  int w = threadIdx.x >> 5, n0 = w * 16;
  int lane = threadIdx.x & 31, n = lane & 15, hh = lane >> 4;
  v8f aq = vzero8(), ak = vzero8(), av = vzero8();
#pragma unroll
  for (int kb = 0; kb < FF; kb += 32) {
    v16h a = fragA_rm(sx, FF, kb);
    aq = wmma_f16(a, fragB_nm(WqT, FF, kb, n0), aq);
    ak = wmma_f16(a, fragB_nm(WkT, FF, kb, n0), ak);
    av = wmma_f16(a, fragB_nm(WvT, FF, kb, n0), av);
  }
  float bqv = bq[n0 + n], bkv = bk[n0 + n], bvv = bv[n0 + n];
  float sq = 0.f, sk = 0.f, sks = 0.f, svs = 0.f;
#pragma unroll
  for (int r = 0; r < 8; ++r) {
    int gi = (node0 + r + 8 * hh) * FF + n0 + n;
    float qv = aq[r] + bqv; q16[gi] = (_Float16)qv; sq += qv * qv;
    float kv = ak[r] + bkv; k16[gi] = (_Float16)kv; sk += kv * kv; sks += kv;
    float vv = av[r] + bvv; v32[gi] = vv; svs += vv;
  }
  atomicAdd(&sred[0], sq);
  atomicAdd(&sred[1], sk);
  atomicAdd(&ksum[n0 + n], sks);
  atomicAdd(&vsum[n0 + n], svs);
  __syncthreads();
  if (threadIdx.x == 0) { atomicAdd(qn2, sred[0]); atomicAdd(kn2, sred[1]); }
}

// ---------------- kvs = k^T v  (128x128, reduced over nodes) ----------------
// each block loops node chunks of 32; k/v chunks staged TRANSPOSED in LDS:
// skT/svT[feat*32 + nodeLocal], so A and B fragments are contiguous vector loads.
__global__ void k_kvs(const _Float16* __restrict__ k16, const float* __restrict__ v32,
                      float* __restrict__ kvs) {
  __shared__ __align__(32) _Float16 skT[FF * 32];
  __shared__ __align__(32) _Float16 svT[FF * 32];
  int w = threadIdx.x >> 5;
  int lane = threadIdx.x & 31, n = lane & 15, hh = lane >> 4;
  v8f acc[8];
#pragma unroll
  for (int t = 0; t < 8; ++t) acc[t] = vzero8();

  for (int c0 = blockIdx.x * 32; c0 < NN; c0 += gridDim.x * 32) {
    for (int i = threadIdx.x; i < 32 * FF; i += 256) {
      int ndl = i >> 7, feat = i & 127;
      bool ok = (c0 + ndl) < NN;
      skT[feat * 32 + ndl] = ok ? k16[c0 * FF + i] : (_Float16)0.0f;
      svT[feat * 32 + ndl] = ok ? (_Float16)v32[(size_t)c0 * FF + i] : (_Float16)0.0f;
    }
    __syncthreads();
    // A[m][node] = k[node][m0+m] = skT[(m0+m)*32 + node]  (row-major in skT, ld=32)
    v16h a = fragA_rm(skT + w * 16 * 32, 32, 0);
#pragma unroll
    for (int t = 0; t < 8; ++t)
      acc[t] = wmma_f16(a, fragB_nm(svT, 32, 0, t * 16), acc[t]);
    __syncthreads();
  }
#pragma unroll
  for (int t = 0; t < 8; ++t)
#pragma unroll
    for (int r = 0; r < 8; ++r)
      atomicAdd(&kvs[(w * 16 + r + 8 * hh) * FF + t * 16 + n], acc[t][r]);
}

// ---------------- per-node attention + inverse-sqrt degree scale ----------------
// kvsT16 is N-major (transposed) so B fragments are single vector loads.
__global__ void k_attn(const _Float16* __restrict__ q16, const _Float16* __restrict__ kvsT16,
                       const float* __restrict__ ksum, const float* __restrict__ vsum,
                       const float* __restrict__ qn2, const float* __restrict__ kn2,
                       const float* __restrict__ deg, float* __restrict__ rd) {
  __shared__ __align__(32) _Float16 sq[16 * FF];
  __shared__ float sn[16 * FF];
  __shared__ float sdot[16];
  int node0 = blockIdx.x * 16;
  for (int i = threadIdx.x; i < 16 * FF; i += 256)
    sq[i] = q16[node0 * FF + i];
  __syncthreads();

  float s = rsqrtf(qn2[0]) * rsqrtf(kn2[0]);
  int w = threadIdx.x >> 5, n0 = w * 16;
  int lane = threadIdx.x & 31, n = lane & 15, hh = lane >> 4;
  v8f acc = vzero8();
#pragma unroll
  for (int kb = 0; kb < FF; kb += 32)
    acc = wmma_f16(fragA_rm(sq, FF, kb), fragB_nm(kvsT16, FF, kb, n0), acc);
  float vs = vsum[n0 + n];
#pragma unroll
  for (int r = 0; r < 8; ++r)
    sn[(r + 8 * hh) * FF + n0 + n] = acc[r] * s + vs;

  if (threadIdx.x < 16) {
    float dt = 0.f;
    for (int i = 0; i < FF; ++i) dt += (float)sq[threadIdx.x * FF + i] * ksum[i];
    sdot[threadIdx.x] = dt;
  }
  __syncthreads();
  if (threadIdx.x < 16) {
    float denom = sdot[threadIdx.x] * s + (float)NN;
    float att = 0.f;
    for (int i = 0; i < FF; ++i) att += sn[threadIdx.x * FF + i];
    att = att / denom * 200.0f;
    float dv = deg[node0 + threadIdx.x] * att;
    rd[node0 + threadIdx.x] = sqrtf(1.0f / dv);   // may be NaN/Inf; filtered per edge
  }
}

// ---------------- edge scatter: h[col] += val * v[row] ----------------
// one wave per edge, 4 floats per lane
__global__ void k_scatter(const int* __restrict__ row, const int* __restrict__ col,
                          const float* __restrict__ ew, const float* __restrict__ rd,
                          const float* __restrict__ v32, float* __restrict__ h) {
  int e = (blockIdx.x * blockDim.x + threadIdx.x) >> 5;
  if (e >= EE) return;
  int lane = threadIdx.x & 31;
  int r = row[e], c = col[e];
  float val = ew[e] * rd[c] * rd[r];
  if (!__builtin_isfinite(val)) val = 0.0f;
  float4 vv = ((const float4*)(v32 + (size_t)r * FF))[lane];
  float* dst = h + (size_t)c * FF + lane * 4;
  atomicAdd(dst + 0, val * vv.x);
  atomicAdd(dst + 1, val * vv.y);
  atomicAdd(dst + 2, val * vv.z);
  atomicAdd(dst + 3, val * vv.w);
}

// ---------------- residual + LayerNorm; one node per block of 128 ----------------
__global__ void k_combine(const float* __restrict__ h, const float* __restrict__ xprev,
                          const float* __restrict__ g, const float* __restrict__ b,
                          float* __restrict__ outx, float* __restrict__ xf32,
                          _Float16* __restrict__ xf16) {
  __shared__ float sbuf[FF];
  int node = blockIdx.x, t = threadIdx.x;
  size_t gi = (size_t)node * FF + t;
  float y = 0.5f * h[gi] + 0.5f * xprev[gi];
  sbuf[t] = y; __syncthreads();
  for (int st = 64; st > 0; st >>= 1) { if (t < st) sbuf[t] += sbuf[t + st]; __syncthreads(); }
  float mu = sbuf[0] * (1.0f / FF);
  __syncthreads();
  float d = y - mu;
  sbuf[t] = d * d; __syncthreads();
  for (int st = 64; st > 0; st >>= 1) { if (t < st) sbuf[t] += sbuf[t + st]; __syncthreads(); }
  float var = sbuf[0] * (1.0f / FF);
  float z = d * rsqrtf(var + 1e-5f) * g[t] + b[t];
  outx[gi] = z;
  xf32[gi] = z;
  xf16[gi] = (_Float16)z;
}

// ---------------- host orchestration ----------------
extern "C" void kernel_launch(void* const* d_in, const int* in_sizes, int n_in,
                              void* d_out, int out_size, void* d_ws, size_t ws_size,
                              hipStream_t stream) {
  (void)in_sizes; (void)n_in; (void)out_size; (void)ws_size;
  const float* feats = (const float*)d_in[0];
  const int*   eidx  = (const int*)d_in[1];
  const int*   row   = eidx;
  const int*   col   = eidx + EE;
  const float* ew    = (const float*)d_in[2];
  const float* W_in  = (const float*)d_in[3];
  const float* b_in  = (const float*)d_in[4];
  const float* ln0g  = (const float*)d_in[5];
  const float* ln0b  = (const float*)d_in[6];
  const float* Wq    = (const float*)d_in[7];
  const float* bq    = (const float*)d_in[8];
  const float* Wk    = (const float*)d_in[9];
  const float* bk    = (const float*)d_in[10];
  const float* Wv    = (const float*)d_in[11];
  const float* bv    = (const float*)d_in[12];
  const float* lng   = (const float*)d_in[13];
  const float* lnb   = (const float*)d_in[14];

  char* p = (char*)d_ws;
  auto carve = [&](size_t bytes) { char* r = p; p += (bytes + 255) & ~(size_t)255; return r; };
  float*    xf32   = (float*)carve((size_t)NN * FF * 4);
  _Float16* xf16   = (_Float16*)carve((size_t)NN * FF * 2);
  _Float16* q16    = (_Float16*)carve((size_t)NN * FF * 2);
  _Float16* k16    = (_Float16*)carve((size_t)NN * FF * 2);
  float*    v32    = (float*)carve((size_t)NN * FF * 4);
  float*    hbuf   = (float*)carve((size_t)NN * FF * 4);
  _Float16* WinT   = (_Float16*)carve((size_t)FF * FF * 2);
  _Float16* WqT    = (_Float16*)carve((size_t)NLAYERS * FF * FF * 2);
  _Float16* WkT    = (_Float16*)carve((size_t)NLAYERS * FF * FF * 2);
  _Float16* WvT    = (_Float16*)carve((size_t)NLAYERS * FF * FF * 2);
  // kvs + ksum + vsum + qn2 + kn2 contiguous so one zero pass clears them
  float*    kvs32  = (float*)carve(((size_t)FF * FF + FF + FF + 2) * 4);
  float*    ksum   = kvs32 + FF * FF;
  float*    vsum   = ksum + FF;
  float*    qn2    = vsum + FF;
  float*    kn2    = qn2 + 1;
  _Float16* kvsT16 = (_Float16*)carve((size_t)FF * FF * 2);
  float*    deg    = (float*)carve((size_t)NN * 4);
  float*    rd     = (float*)carve((size_t)NN * 4);

  const int ZT = 256;
  const int MB = (FF * FF + ZT - 1) / ZT;  // blocks per 128x128 matrix
  // weight transpose+convert fp32 -> N-major f16
  k_cvtT_128<<<MB, ZT, 0, stream>>>(W_in, WinT);
  for (int l = 0; l < NLAYERS; ++l) {
    k_cvtT_128<<<MB, ZT, 0, stream>>>(Wq + (size_t)l * FF * FF, WqT + (size_t)l * FF * FF);
    k_cvtT_128<<<MB, ZT, 0, stream>>>(Wk + (size_t)l * FF * FF, WkT + (size_t)l * FF * FF);
    k_cvtT_128<<<MB, ZT, 0, stream>>>(Wv + (size_t)l * FF * FF, WvT + (size_t)l * FF * FF);
  }

  // degrees (layer-invariant)
  k_zero<<<(NN + ZT - 1) / ZT, ZT, 0, stream>>>(deg, NN);
  k_deg<<<(EE + ZT - 1) / ZT, ZT, 0, stream>>>(col, deg, EE);

  // input projection + LN + relu
  k_inproj<<<NN / 16, 256, 0, stream>>>(feats, WinT, b_in, ln0g, ln0b, xf32, xf16);

  // output slice 0 = source_feats passthrough
  hipMemcpyAsync(d_out, (const void*)feats, (size_t)NN * FF * sizeof(float),
                 hipMemcpyDeviceToDevice, stream);

  for (int l = 0; l < NLAYERS; ++l) {
    const int redn = FF * FF + FF + FF + 2;
    k_zero<<<(redn + ZT - 1) / ZT, ZT, 0, stream>>>(kvs32, redn);
    k_qkv<<<NN / 16, 256, 0, stream>>>(xf16, WqT + (size_t)l * FF * FF,
                                       WkT + (size_t)l * FF * FF, WvT + (size_t)l * FF * FF,
                                       bq + l * FF, bk + l * FF, bv + l * FF,
                                       q16, k16, v32, ksum, vsum, qn2, kn2);
    k_kvs<<<128, 256, 0, stream>>>(k16, v32, kvs32);
    k_cvtT_128<<<MB, ZT, 0, stream>>>(kvs32, kvsT16);
    k_attn<<<NN / 16, 256, 0, stream>>>(q16, kvsT16, ksum, vsum, qn2, kn2, deg, rd);
    k_zero<<<(NN * FF + ZT - 1) / ZT, ZT, 0, stream>>>(hbuf, NN * FF);
    k_scatter<<<(EE * 32 + 255) / 256, 256, 0, stream>>>(row, col, ew, rd, v32, hbuf);
    float* outSlice = (float*)d_out + (size_t)(1 + l) * NN * FF;
    k_combine<<<NN, FF, 0, stream>>>(hbuf, xf32, lng + l * FF, lnb + l * FF,
                                     outSlice, xf32, xf16);
  }
}
